// LinearAttention_80453327389548
// MI455X (gfx1250) — matrix operand order repaired and checked by reading, not compiled, not run on hardware
//
#include <hip/hip_runtime.h>
#include <cstdint>

// ---------------------------------------------------------------------------
// LinearAttention fused pipeline for gfx1250 (MI455X), bf16 WMMA + fp32 accum
// ---------------------------------------------------------------------------

typedef __bf16 bf16;
typedef __bf16 bf16x16 __attribute__((ext_vector_type(16)));
typedef float  f32x8   __attribute__((ext_vector_type(8)));
typedef int    v4i     __attribute__((vector_size(16)));   // matches builtin's V4i

#define HEADS 8
#define DHEAD 64
#define CDIM 256     // channels
#define HID 512      // HEADS*DHEAD
#define NSP 4096     // spatial (64*64)
#define BATCH 8
#define KCHUNKS 16   // split-K factor for context GEMM

#if __has_builtin(__builtin_amdgcn_global_load_async_to_lds_b128) && \
    __has_builtin(__builtin_amdgcn_s_wait_asynccnt)
#define HAS_ASYNC_LDS 1
#else
#define HAS_ASYNC_LDS 0
#endif

static __device__ __forceinline__ f32x8 wmma_bf16(bf16x16 a, bf16x16 b, f32x8 c) {
    return __builtin_amdgcn_wmma_f32_16x16x32_bf16(
        /*neg_a=*/false, a, /*neg_b=*/false, b,
        /*c_mod=*/(short)0, c, /*reuse_a=*/false, /*reuse_b=*/false);
}

#if HAS_ASYNC_LDS
typedef __attribute__((address_space(1))) v4i* gptr_v4i;
typedef __attribute__((address_space(3))) v4i* lptr_v4i;
// 16-byte async global->LDS copy per lane (tracked by ASYNCcnt).
static __device__ __forceinline__ void async_copy16(const void* g, void* l) {
    __builtin_amdgcn_global_load_async_to_lds_b128(
        (gptr_v4i)(uintptr_t)g,
        (lptr_v4i)(uint32_t)(uintptr_t)l,
        0, 0);
}
#endif

// A fragment: 16(M) x 32(K) bf16, tile in LDS row-major [m][k], row stride ld.
// Lane layout (wave32): lanes 0-15 -> M=lane, K in {0..7}U{16..23};
//                       lanes 16-31 -> M=lane-16, K in {8..15}U{24..31}.
static __device__ __forceinline__ bf16x16 frag_a(const bf16* tile, int lane, int ld) {
    const int half = lane >> 4, m = lane & 15;
    const bf16* r = tile + m * ld;
    bf16x16 a;
#pragma unroll
    for (int e = 0; e < 8; ++e) a[e] = r[half * 8 + e];
#pragma unroll
    for (int e = 0; e < 8; ++e) a[8 + e] = r[16 + half * 8 + e];
    return a;
}

// B fragment: 32(K) x 16(N) bf16. Tile stored TRANSPOSED in LDS as [n][k],
// row stride ld. Lanes 0-15 -> N=lane, K in {0..7}U{16..23}; lanes 16-31 -> N=lane-16, K in {8..15}U{24..31}.
static __device__ __forceinline__ bf16x16 frag_b(const bf16* tile, int lane, int ld) {
    const int half = lane >> 4, n = lane & 15;
    const bf16* r = tile + n * ld;
    bf16x16 b;
#pragma unroll
    for (int e = 0; e < 8; ++e) b[e] = r[half * 8 + e];
#pragma unroll
    for (int e = 0; e < 8; ++e) b[8 + e] = r[16 + half * 8 + e];
    return b;
}

static __device__ __forceinline__ f32x8 zero8() {
    f32x8 z = {0.f, 0.f, 0.f, 0.f, 0.f, 0.f, 0.f, 0.f};
    return z;
}

// ---------------------------------------------------------------------------
// K1: te = silu(time_emb) @ w_mlp^T + b_mlp  -> ss[b][512] (scale | shift)
// ---------------------------------------------------------------------------
__global__ void la_time_mlp(const float* __restrict__ time_emb,
                            const float* __restrict__ w_mlp,
                            const float* __restrict__ b_mlp,
                            float* __restrict__ ss) {
    const int gid = blockIdx.x * 256 + threadIdx.x;   // 8*512 = 4096
    const int b = gid >> 9, j = gid & 511;
    const float* te = time_emb + b * 512;
    const float* w  = w_mlp + j * 512;
    float s = b_mlp[j];
    for (int i = 0; i < 512; ++i) {
        float v = te[i];
        s += (v / (1.0f + __expf(-v))) * w[i];        // silu
    }
    ss[gid] = s;
}

// ---------------------------------------------------------------------------
// K2: qkv[b][o][p] = sum_c w_qkv[o][c] * (x[b][c][p]*(1+scale)+shift), bf16 out
//     block tile: M=128 (o), N=64 (p), K=256 in steps of 32; 8 waves
// ---------------------------------------------------------------------------
__global__ void la_qkv_gemm(const float* __restrict__ x,
                            const float* __restrict__ w_qkv,
                            const float* __restrict__ ss,
                            bf16* __restrict__ qkv) {
    const int b  = blockIdx.z;
    const int m0 = blockIdx.y * 128;
    const int n0 = blockIdx.x * 64;
    __shared__ bf16 As[128 * 32];   // [m][k]
    __shared__ bf16 Bs[64 * 32];    // [n][k] (transposed)
    const int t = threadIdx.x, wv = t >> 5, lane = t & 31;
    const float* scale = ss + b * 512;
    const float* shift = scale + 256;

    f32x8 acc[4] = {zero8(), zero8(), zero8(), zero8()};

    for (int k0 = 0; k0 < CDIM; k0 += 32) {
        for (int i = t; i < 128 * 32; i += 256) {
            int m = i >> 5, kk = i & 31;
            As[i] = (bf16)w_qkv[(m0 + m) * CDIM + k0 + kk];
        }
        for (int i = t; i < 64 * 32; i += 256) {
            int kk = i >> 6, n = i & 63;                  // coalesced in n
            int c = k0 + kk;
            float xv = x[((size_t)b * CDIM + c) * NSP + n0 + n];
            Bs[n * 32 + kk] = (bf16)(xv * (scale[c] + 1.0f) + shift[c]);
        }
        __syncthreads();
        bf16x16 a = frag_a(&As[(wv * 16) * 32], lane, 32);
#pragma unroll
        for (int nt = 0; nt < 4; ++nt) {
            bf16x16 bf = frag_b(&Bs[(nt * 16) * 32], lane, 32);
            acc[nt] = wmma_bf16(a, bf, acc[nt]);
        }
        __syncthreads();
    }
    const int half = lane >> 4, nn = lane & 15;
    const int m_base = m0 + wv * 16 + 8 * half;
#pragma unroll
    for (int nt = 0; nt < 4; ++nt)
#pragma unroll
        for (int r = 0; r < 8; ++r)
            qkv[((size_t)b * 1536 + (m_base + r)) * NSP + n0 + nt * 16 + nn] =
                (bf16)acc[nt][r];
}

// ---------------------------------------------------------------------------
// K3: in-place softmax over n=4096 for each k row (b, h*64+d); 1 block/row
// ---------------------------------------------------------------------------
__global__ void la_k_softmax(bf16* __restrict__ qkv) {
    const int idx = blockIdx.x;                 // 8*512 rows
    const int b = idx >> 9, r = idx & 511;
    bf16* row = qkv + ((size_t)b * 1536 + 512 + r) * NSP;
    __shared__ float red[256];
    const int t = threadIdx.x;

    float mx = -1e30f;
    for (int i = t; i < NSP; i += 256) mx = fmaxf(mx, (float)row[i]);
    red[t] = mx; __syncthreads();
    for (int s = 128; s > 0; s >>= 1) { if (t < s) red[t] = fmaxf(red[t], red[t + s]); __syncthreads(); }
    mx = red[0]; __syncthreads();

    float sm = 0.f;
    for (int i = t; i < NSP; i += 256) sm += __expf((float)row[i] - mx);
    red[t] = sm; __syncthreads();
    for (int s = 128; s > 0; s >>= 1) { if (t < s) red[t] += red[t + s]; __syncthreads(); }
    const float inv = 1.0f / red[0];
    __syncthreads();

    for (int i = t; i < NSP; i += 256)
        row[i] = (bf16)(__expf((float)row[i] - mx) * inv);
}

// ---------------------------------------------------------------------------
// K4: split-K context GEMM. ctx_part[chunk][bh][d][e] = sum_{n in chunk}
//     ksm[d][n]*v[e][n]. One block per (chunk, bh): M=64(d), N=64(e), K=256.
//     Pure byte-copy tiles -> async global->LDS + double buffering when
//     available (ASYNCcnt path), else synchronous fallback.
// ---------------------------------------------------------------------------
__global__ void la_context_gemm(const bf16* __restrict__ qkv,
                                float* __restrict__ ctx_part) {
    const int bh = blockIdx.y, b = bh >> 3, h = bh & 7;
    const int chunk = blockIdx.x;
    const int kbase = chunk * 256;
    const bf16* Kp = qkv + ((size_t)b * 1536 + 512 + h * 64) * NSP;
    const bf16* Vp = qkv + ((size_t)b * 1536 + 1024 + h * 64) * NSP;
    const int t = threadIdx.x, wv = t >> 5, lane = t & 31;
    const int mt = wv & 3, ng = (wv >> 2) * 2;

    f32x8 acc[2] = {zero8(), zero8()};

#if HAS_ASYNC_LDS
    __shared__ bf16 As[2][64 * 32];   // ksm [d][k]
    __shared__ bf16 Bs[2][64 * 32];   // v   [e][k]
    // per-thread 16B copy: row = t>>2 (0..63), seg = (t&3)*8 bf16
    const int crow = t >> 2, cseg = (t & 3) * 8;

    // preload tile 0
    async_copy16(Kp + (size_t)crow * NSP + kbase + cseg, &As[0][crow * 32 + cseg]);
    async_copy16(Vp + (size_t)crow * NSP + kbase + cseg, &Bs[0][crow * 32 + cseg]);

    for (int it = 0; it < 8; ++it) {
        if (it < 7) {
            const int k1 = kbase + (it + 1) * 32;
            const int nb = (it + 1) & 1;
            async_copy16(Kp + (size_t)crow * NSP + k1 + cseg, &As[nb][crow * 32 + cseg]);
            async_copy16(Vp + (size_t)crow * NSP + k1 + cseg, &Bs[nb][crow * 32 + cseg]);
            __builtin_amdgcn_s_wait_asynccnt(2);   // tile `it` landed
        } else {
            __builtin_amdgcn_s_wait_asynccnt(0);
        }
        __syncthreads();
        const bf16* Atile = As[it & 1];
        const bf16* Btile = Bs[it & 1];
        bf16x16 a = frag_a(&Atile[(mt * 16) * 32], lane, 32);
#pragma unroll
        for (int j = 0; j < 2; ++j) {
            bf16x16 bf = frag_b(&Btile[((ng + j) * 16) * 32], lane, 32);
            acc[j] = wmma_bf16(a, bf, acc[j]);
        }
        __syncthreads();   // done reading buf (it&1) before it is refilled
    }
#else
    __shared__ bf16 As[64 * 32];
    __shared__ bf16 Bs[64 * 32];
    for (int it = 0; it < 8; ++it) {
        const int k0 = kbase + it * 32;
        for (int i = t; i < 64 * 32; i += 256) {
            int d = i >> 5, kk = i & 31;
            As[i] = Kp[(size_t)d * NSP + k0 + kk];
            Bs[i] = Vp[(size_t)d * NSP + k0 + kk];
        }
        __syncthreads();
        bf16x16 a = frag_a(&As[(mt * 16) * 32], lane, 32);
#pragma unroll
        for (int j = 0; j < 2; ++j) {
            bf16x16 bf = frag_b(&Bs[((ng + j) * 16) * 32], lane, 32);
            acc[j] = wmma_bf16(a, bf, acc[j]);
        }
        __syncthreads();
    }
#endif

    float* C = ctx_part + ((size_t)chunk * 64 + bh) * (64 * 64);
    const int half = lane >> 4, nn = lane & 15;
#pragma unroll
    for (int j = 0; j < 2; ++j)
#pragma unroll
        for (int r = 0; r < 8; ++r) {
            int d = mt * 16 + 8 * half + r;
            int e = (ng + j) * 16 + nn;
            C[d * 64 + e] = acc[j][r];
        }
}

// ---------------------------------------------------------------------------
// K4b: ctx[bh][d][e] = (1/NSP) * sum_chunk ctx_part[chunk][bh][d][e]
// ---------------------------------------------------------------------------
__global__ void la_ctx_reduce(const float* __restrict__ ctx_part,
                              float* __restrict__ ctx) {
    const int i = blockIdx.x * 256 + threadIdx.x;     // 64*64*64 = 262144
    float s = 0.f;
#pragma unroll
    for (int c = 0; c < KCHUNKS; ++c)
        s += ctx_part[(size_t)c * (64 * 64 * 64) + i];
    ctx[i] = s * (1.0f / (float)NSP);                 // fold v/n scaling
}

// ---------------------------------------------------------------------------
// K5: column-softmax(q over d)*d^-0.5, then attn[e][n] = sum_d ctx[d][e]*qsm[d][n]
//     one block per (b, h, 128-wide n tile); M=64 (e), K=64 (d), N=128
// ---------------------------------------------------------------------------
__global__ void la_attn_gemm(const bf16* __restrict__ qkv,
                             const float* __restrict__ ctx,
                             bf16* __restrict__ attn) {
    const int b = blockIdx.z, h = blockIdx.y, n0 = blockIdx.x * 128;
    __shared__ float Qs[64 * 128];   // raw q, fp32, [d][n]
    __shared__ bf16 As[64 * 64];     // A[m=e][k=d] = ctx[d][e]
    __shared__ bf16 Bs[128 * 64];    // B[n][k=d]   = qsm[d][n] * 0.125
    const int t = threadIdx.x;

    const bf16* Q = qkv + ((size_t)b * 1536 + h * 64) * NSP;
    for (int i = t; i < 64 * 128; i += 256) {
        int d = i >> 7, n = i & 127;
        Qs[i] = (float)Q[(size_t)d * NSP + n0 + n];
    }
    const float* C = ctx + (size_t)(b * 8 + h) * (64 * 64);
    for (int i = t; i < 64 * 64; i += 256) {
        int e = i >> 6, d = i & 63;
        As[i] = (bf16)C[d * 64 + e];
    }
    __syncthreads();

    if (t < 128) {                                  // per-column softmax over d
        float mx = -1e30f;
        for (int d = 0; d < 64; ++d) mx = fmaxf(mx, Qs[d * 128 + t]);
        float sm = 0.f;
        for (int d = 0; d < 64; ++d) sm += __expf(Qs[d * 128 + t] - mx);
        const float inv = 0.125f / sm;              // DIM_HEAD^-0.5 = 0.125
        for (int d = 0; d < 64; ++d)
            Bs[t * 64 + d] = (bf16)(__expf(Qs[d * 128 + t] - mx) * inv);
    }
    __syncthreads();

    const int wv = t >> 5, lane = t & 31;
    const int mt = wv & 3, nb = (wv >> 2) * 4;
    f32x8 acc[4] = {zero8(), zero8(), zero8(), zero8()};
#pragma unroll
    for (int k0 = 0; k0 < 64; k0 += 32) {
        bf16x16 a = frag_a(&As[(mt * 16) * 64 + k0], lane, 64);
#pragma unroll
        for (int j = 0; j < 4; ++j) {
            bf16x16 bf = frag_b(&Bs[((nb + j) * 16) * 64 + k0], lane, 64);
            acc[j] = wmma_bf16(a, bf, acc[j]);
        }
    }
    const int half = lane >> 4, nn = lane & 15;
    const int e_base = mt * 16 + 8 * half;
#pragma unroll
    for (int j = 0; j < 4; ++j)
#pragma unroll
        for (int r = 0; r < 8; ++r)
            attn[((size_t)b * HID + h * 64 + e_base + r) * NSP + n0 + (nb + j) * 16 + nn] =
                (bf16)acc[j][r];
}

// ---------------------------------------------------------------------------
// K6: out = LN_c( w_out @ attn + b_out ) * g_ln ; block covers full M=256,
//     N tile = 32, K=512 in steps of 32 -> LayerNorm fused in LDS
// ---------------------------------------------------------------------------
__global__ void la_out_gemm_ln(const bf16* __restrict__ attn,
                               const float* __restrict__ w_out,
                               const float* __restrict__ b_out,
                               const float* __restrict__ g_ln,
                               float* __restrict__ out) {
    const int b = blockIdx.y, n0 = blockIdx.x * 32;
    __shared__ bf16 As[256 * 32];    // [m][k]
    __shared__ bf16 Bs[32 * 32];     // [n][k]
    __shared__ float Obuf[256 * 32]; // fp32 pre-LN
    __shared__ float mean_s[32], rstd_s[32];
    const int t = threadIdx.x, wv = t >> 5, lane = t & 31;

    f32x8 acc[2][2] = {{zero8(), zero8()}, {zero8(), zero8()}};

    for (int k0 = 0; k0 < HID; k0 += 32) {
        for (int i = t; i < 256 * 32; i += 256) {
            int m = i >> 5, kk = i & 31;
            As[i] = (bf16)w_out[m * HID + k0 + kk];
        }
        for (int i = t; i < 32 * 32; i += 256) {
            int kk = i >> 5, n = i & 31;                 // coalesced in n
            Bs[n * 32 + kk] = attn[((size_t)b * HID + k0 + kk) * NSP + n0 + n];
        }
        __syncthreads();
#pragma unroll
        for (int mi = 0; mi < 2; ++mi) {
            bf16x16 a = frag_a(&As[((wv * 2 + mi) * 16) * 32], lane, 32);
#pragma unroll
            for (int nj = 0; nj < 2; ++nj) {
                bf16x16 bf = frag_b(&Bs[(nj * 16) * 32], lane, 32);
                acc[mi][nj] = wmma_bf16(a, bf, acc[mi][nj]);
            }
        }
        __syncthreads();
    }

    const int half = lane >> 4, nn = lane & 15;
#pragma unroll
    for (int mi = 0; mi < 2; ++mi) {
        const int m_base = (wv * 2 + mi) * 16 + 8 * half;
#pragma unroll
        for (int nj = 0; nj < 2; ++nj)
#pragma unroll
            for (int r = 0; r < 8; ++r)
                Obuf[(m_base + r) * 32 + nj * 16 + nn] = acc[mi][nj][r] + b_out[m_base + r];
    }
    __syncthreads();

    if (t < 32) {
        float s = 0.f, s2 = 0.f;
        for (int m = 0; m < 256; ++m) { float v = Obuf[m * 32 + t]; s += v; s2 += v * v; }
        const float mean = s * (1.0f / 256.0f);
        const float var  = s2 * (1.0f / 256.0f) - mean * mean;
        mean_s[t] = mean;
        rstd_s[t] = rsqrtf(var + 1e-5f);
    }
    __syncthreads();

    for (int i = t; i < 256 * 32; i += 256) {
        int m = i >> 5, n = i & 31;
        out[((size_t)b * CDIM + m) * NSP + n0 + n] =
            (Obuf[i] - mean_s[n]) * rstd_s[n] * g_ln[m];
    }
}

// ---------------------------------------------------------------------------
// Launcher
// ---------------------------------------------------------------------------
extern "C" void kernel_launch(void* const* d_in, const int* in_sizes, int n_in,
                              void* d_out, int out_size, void* d_ws, size_t ws_size,
                              hipStream_t stream) {
    (void)in_sizes; (void)n_in; (void)out_size; (void)ws_size;
    const float* x        = (const float*)d_in[0];   // [8,256,64,64]
    const float* time_emb = (const float*)d_in[1];   // [8,512]
    const float* w_qkv    = (const float*)d_in[2];   // [1536,256]
    const float* w_out    = (const float*)d_in[3];   // [256,512]
    const float* b_out    = (const float*)d_in[4];   // [256]
    const float* g_ln     = (const float*)d_in[5];   // [256]
    const float* w_mlp    = (const float*)d_in[6];   // [512,512]
    const float* b_mlp    = (const float*)d_in[7];   // [512]
    float* out = (float*)d_out;

    // workspace layout (bytes)
    char* ws = (char*)d_ws;
    size_t off = 0;
    float* ss = (float*)(ws + off);          off += (size_t)BATCH * 512 * 4;          // 16 KB
    bf16* qkv = (bf16*)(ws + off);           off += (size_t)BATCH * 1536 * NSP * 2;   // 100.7 MB
    float* ctx_part = (float*)(ws + off);    off += (size_t)KCHUNKS * 64 * 4096 * 4;  // 16.8 MB
    float* ctx = (float*)(ws + off);         off += (size_t)64 * 4096 * 4;            // 1 MB
    bf16* attn = (bf16*)(ws + off);          off += (size_t)BATCH * HID * NSP * 2;    // 33.6 MB

    la_time_mlp<<<dim3(16), dim3(256), 0, stream>>>(time_emb, w_mlp, b_mlp, ss);
    la_qkv_gemm<<<dim3(64, 12, 8), dim3(256), 0, stream>>>(x, w_qkv, ss, qkv);
    la_k_softmax<<<dim3(4096), dim3(256), 0, stream>>>(qkv);
    la_context_gemm<<<dim3(KCHUNKS, 64), dim3(256), 0, stream>>>(qkv, ctx_part);
    la_ctx_reduce<<<dim3(1024), dim3(256), 0, stream>>>(ctx_part, ctx);
    la_attn_gemm<<<dim3(32, 8, 8), dim3(256), 0, stream>>>(qkv, ctx, attn);
    la_out_gemm_ln<<<dim3(128, 8), dim3(256), 0, stream>>>(attn, w_out, b_out, g_ln, out);
}
